// SqueezedSelfAttentionLayer_25872882991932
// MI455X (gfx1250) — matrix operand order repaired and checked by reading, not compiled, not run on hardware
//
#include <hip/hip_runtime.h>
#include <hip/hip_bf16.h>

#define NN 4096
#define EE 256
#define HH 8
#define DH 32

typedef _Float16 v8h  __attribute__((ext_vector_type(8)));
typedef _Float16 v16h __attribute__((ext_vector_type(16)));
typedef float    v8f  __attribute__((ext_vector_type(8)));

#define NEG_INF (-3.0e38f)

// ---------------------------------------------------------------------------
// Fragment helpers (wave32, 16x16x32 f16 WMMA)
//
// A-matrix (16x32, f16): lane L -> row M = L&15, half = L>>4.
//   v16h elements 0..7  hold K = 8*half + {0..7}
//   v16h elements 8..15 hold K = 16 + 8*half + {0..7}
// B-matrix (32x16) symmetric: lane L -> col N = L&15, same K mapping.
// C/D (16x16 f32, 8 VGPRs): lane L -> col N = L&15; reg r -> row M = r + 8*(L>>4)
// ---------------------------------------------------------------------------
__device__ __forceinline__ v16h load_frag_rm(const _Float16* rowp, int hf) {
  const _Float16* p = rowp + 8 * hf;
  v8h lo = *(const v8h*)p;          // K = k0 + 8*half + 0..7   (16 bytes)
  v8h hi = *(const v8h*)(p + 16);   // K = k0 + 16 + 8*half + 0..7
  return __builtin_shufflevector(lo, hi, 0,1,2,3,4,5,6,7,8,9,10,11,12,13,14,15);
}

__device__ __forceinline__ v16h load_frag_cm(const _Float16* base, int ld,
                                             int k0, int colo, int hf) {
  v16h r;
#pragma unroll
  for (int e = 0; e < 8; ++e)
    r[e] = base[(size_t)(k0 + 8 * hf + e) * ld + colo];
#pragma unroll
  for (int e = 0; e < 8; ++e)
    r[8 + e] = base[(size_t)(k0 + 16 + 8 * hf + e) * ld + colo];
  return r;
}

__device__ __forceinline__ v8f wmma16(v16h a, v16h b, v8f c) {
  return __builtin_amdgcn_wmma_f32_16x16x32_f16(
      /*neg_a=*/false, a, /*neg_b=*/false, b,
      /*c_mod=*/(short)0, c, /*reuse_a=*/false, /*reuse_b=*/false);
}

__device__ __forceinline__ float rmax16(float v) {
#pragma unroll
  for (int m = 1; m < 16; m <<= 1) v = fmaxf(v, __shfl_xor(v, m, 32));
  return v;
}
__device__ __forceinline__ float rsum16(float v) {
#pragma unroll
  for (int m = 1; m < 16; m <<= 1) v += __shfl_xor(v, m, 32);
  return v;
}

// ---------------------------------------------------------------------------
// CDNA5 async global->LDS copy (ASYNCcnt path; not covered by builtins)
//   LDS byte address taken from the lds_off VGPR;  16 bytes per lane per issue
// ---------------------------------------------------------------------------
__device__ __forceinline__ void async_ld_b128(unsigned lds_off,
                                              unsigned long long gaddr) {
  asm volatile("global_load_async_to_lds_b128 %0, %1, off"
               :: "v"(lds_off), "v"(gaddr)
               : "memory");
}
__device__ __forceinline__ void wait_async_all() {
  asm volatile("s_wait_asynccnt 0x0" ::: "memory");
}
__device__ __forceinline__ unsigned lds_off_of(const void* p) {
  // generic shared pointer: low 32 bits are the LDS byte address
  return (unsigned)(uintptr_t)p;
}

// ---------------------------------------------------------------------------
// Kernel 0: f32 -> f16 staging (xp = x + pe, x, and the 4 weight matrices)
// ---------------------------------------------------------------------------
__global__ __launch_bounds__(256) void prep_kernel(
    const float* __restrict__ x, const float* __restrict__ pe,
    const float* __restrict__ Wq, const float* __restrict__ Wk,
    const float* __restrict__ Wv, const float* __restrict__ Wo,
    _Float16* __restrict__ xp16, _Float16* __restrict__ x16,
    _Float16* __restrict__ Wq16, _Float16* __restrict__ Wk16,
    _Float16* __restrict__ Wv16, _Float16* __restrict__ Wo16) {
  int i = blockIdx.x * 256 + threadIdx.x;
  const int NE = NN * EE;
  if (i < NE) {
    float xv = x[i];
    xp16[i] = (_Float16)(xv + pe[i]);
    x16[i]  = (_Float16)xv;
  } else {
    int j = i - NE;
    if (j < 4 * EE * EE) {
      int w = j >> 16;          // EE*EE == 65536
      int q = j & 65535;
      const float* src = (w == 0) ? Wq : (w == 1) ? Wk : (w == 2) ? Wv : Wo;
      _Float16*    dst = (w == 0) ? Wq16 : (w == 1) ? Wk16 : (w == 2) ? Wv16 : Wo16;
      dst[q] = (_Float16)src[q];
    }
  }
}

// ---------------------------------------------------------------------------
// Kernel 1: QKV projections.  grid.x = N/16 row tiles, 8 waves/block.
// A-fragments (xp and x, 8 k-steps each) are loaded ONCE into registers and
// reused across all 48 WMMAs per wave.  p is a compile-time unrolled loop so
// the register-array selection is static.
// ---------------------------------------------------------------------------
__global__ __launch_bounds__(256) void proj_kernel(
    const _Float16* __restrict__ xp16, const _Float16* __restrict__ x16,
    const _Float16* __restrict__ Wq16, const _Float16* __restrict__ Wk16,
    const _Float16* __restrict__ Wv16,
    const float* __restrict__ bq, const float* __restrict__ bk,
    const float* __restrict__ bv,
    _Float16* __restrict__ Q16, _Float16* __restrict__ K16,
    _Float16* __restrict__ V16) {
  const int m0  = blockIdx.x * 16;
  const int wv  = threadIdx.x >> 5;
  const int lane = threadIdx.x & 31;
  const int col = lane & 15, hf = lane >> 4;

  v16h axp[8], ax[8];
  {
    const _Float16* xprow = xp16 + (size_t)(m0 + col) * EE;
    const _Float16* xrow  = x16  + (size_t)(m0 + col) * EE;
#pragma unroll
    for (int k = 0; k < 8; ++k) {
      axp[k] = load_frag_rm(xprow + 32 * k, hf);
      ax[k]  = load_frag_rm(xrow  + 32 * k, hf);
    }
  }

#pragma unroll
  for (int p = 0; p < 3; ++p) {                 // 0=Q 1=K 2=V (compile-time)
    const _Float16* W    = (p == 0) ? Wq16 : (p == 1) ? Wk16 : Wv16;
    const float*    bias = (p == 0) ? bq : (p == 1) ? bk : bv;
    _Float16*       dst  = (p == 0) ? Q16 : (p == 1) ? K16 : V16;

#pragma unroll
    for (int t = 0; t < 2; ++t) {               // 8 waves x 2 tiles = 16 n-tiles
      int nt = wv * 2 + t;
      const _Float16* wrow = W + (size_t)(nt * 16 + col) * EE;
      v8f acc = {};
#pragma unroll
      for (int k = 0; k < 8; ++k) {
        v16h af = (p == 2) ? ax[k] : axp[k];
        acc = wmma16(af, load_frag_rm(wrow + 32 * k, hf), acc);
      }
      int eg = nt * 16 + col;
      float bb = bias[eg];
      int h = eg >> 5, d = eg & 31;
#pragma unroll
      for (int r = 0; r < 8; ++r) {
        int m = m0 + r + 8 * hf;
        dst[((size_t)h * NN + m) * DH + d] = (_Float16)(acc[r] + bb);
      }
    }
  }
}

// ---------------------------------------------------------------------------
// Kernel 2: block-diagonal flash attention with double-buffered async
// global->LDS staging of K/V tiles (overlapped with WMMA compute).
// grid = (N/16 row-tiles, H heads), one wave32 per block.
// ---------------------------------------------------------------------------
__global__ __launch_bounds__(32) void attn_kernel(
    const _Float16* __restrict__ Q16, const _Float16* __restrict__ K16,
    const _Float16* __restrict__ V16, const long long* __restrict__ xb,
    _Float16* __restrict__ ctx16) {
  __shared__ alignas(16) _Float16 Kt[2][32][32];   // 4 KB
  __shared__ alignas(16) _Float16 Vt[2][32][32];   // 4 KB
  __shared__ alignas(16) _Float16 P[16][32];       // 1 KB

  const int m0 = blockIdx.x * 16;
  const int h  = blockIdx.y;
  const int lane = threadIdx.x;
  const int col = lane & 15, hf = lane >> 4;

  const _Float16* Qh = Q16 + (size_t)h * NN * DH;
  const _Float16* Kh = K16 + (size_t)h * NN * DH;
  const _Float16* Vh = V16 + (size_t)h * NN * DH;

  // per-row batch ids (rows this lane owns in C layout)
  int brow[8];
#pragma unroll
  for (int r = 0; r < 8; ++r) brow[r] = (int)xb[m0 + r + 8 * hf];

  // contiguous column range covering all batches present in this row tile
  long long bF = xb[m0], bL = xb[m0 + 15];
  int l = 0, rr = m0;
  while (l < rr) { int mid = (l + rr) >> 1; if (xb[mid] <  bF) l = mid + 1; else rr = mid; }
  const int lo = l;
  l = m0 + 15; rr = NN;
  while (l < rr) { int mid = (l + rr) >> 1; if (xb[mid] <= bL) l = mid + 1; else rr = mid; }
  const int hi = l;

  const v16h qfrag = load_frag_rm(Qh + (size_t)(m0 + col) * DH, hf);

  v8f acc0 = {}, acc1 = {};
  float mrun[8], lrun[8];
#pragma unroll
  for (int r = 0; r < 8; ++r) { mrun[r] = NEG_INF; lrun[r] = 0.0f; }

  const float scale = 0.17677669529663687f;   // 1/sqrt(32)

  const int jt0 = lo & ~31;
  const int nit = (hi - jt0 + 31) >> 5;

  // issue async copies of K/V rows [jt, jt+32) into LDS buffer b
  auto stage = [&](int jt, int b) {
    unsigned long long gK = (unsigned long long)(uintptr_t)(Kh + (size_t)(jt + lane) * DH);
    unsigned long long gV = (unsigned long long)(uintptr_t)(Vh + (size_t)(jt + lane) * DH);
    unsigned lK = lds_off_of(&Kt[b][lane][0]);
    unsigned lV = lds_off_of(&Vt[b][lane][0]);
    async_ld_b128(lK,      gK);
    async_ld_b128(lK + 16, gK + 16);
    async_ld_b128(lK + 32, gK + 32);
    async_ld_b128(lK + 48, gK + 48);
    async_ld_b128(lV,      gV);
    async_ld_b128(lV + 16, gV + 16);
    async_ld_b128(lV + 32, gV + 32);
    async_ld_b128(lV + 48, gV + 48);
  };

  stage(jt0, 0);

  for (int it = 0; it < nit; ++it) {
    const int jt = jt0 + it * 32;
    const int b  = it & 1;

    wait_async_all();                 // current buffer resident in LDS
    if (it + 1 < nit) stage(jt + 32, b ^ 1);   // overlap next tile with compute

    // ---- scores: two 16x16 tiles, K = DH = 32 (K-frags from LDS) ----------
    v16h kf0 = load_frag_rm(&Kt[b][col][0], hf);
    v16h kf1 = load_frag_rm(&Kt[b][16 + col][0], hf);
    v8f z = {};
    v8f s0 = wmma16(qfrag, kf0, z);
    v8f s1 = wmma16(qfrag, kf1, z);

    int bc0 = (int)xb[jt + col];
    int bc1 = (int)xb[jt + 16 + col];

    // ---- online softmax (rows replicated across 16 lanes of each half) ----
#pragma unroll
    for (int r = 0; r < 8; ++r) {
      float a = s0[r] * scale, bsc = s1[r] * scale;
      bool ok0 = (bc0 == brow[r]);
      bool ok1 = (bc1 == brow[r]);
      float t = fmaxf(ok0 ? a : NEG_INF, ok1 ? bsc : NEG_INF);
      t = rmax16(t);
      float mnew = fmaxf(mrun[r], t);
      float c    = __expf(mrun[r] - mnew);
      float p0   = ok0 ? __expf(a - mnew) : 0.0f;
      float p1   = ok1 ? __expf(bsc - mnew) : 0.0f;
      lrun[r] = lrun[r] * c + rsum16(p0 + p1);
      mrun[r] = mnew;
      acc0[r] *= c;
      acc1[r] *= c;
      P[r + 8 * hf][col]      = (_Float16)p0;
      P[r + 8 * hf][16 + col] = (_Float16)p1;
    }
    __syncthreads();

    // ---- ctx += P @ V : A-frag from LDS, two d-tiles of V from LDS --------
    v16h pf  = load_frag_rm(&P[col][0], hf);
    v16h vf0 = load_frag_cm(&Vt[b][0][0], 32, 0, col, hf);
    v16h vf1 = load_frag_cm(&Vt[b][0][0], 32, 0, 16 + col, hf);
    acc0 = wmma16(pf, vf0, acc0);
    acc1 = wmma16(pf, vf1, acc1);
    __syncthreads();
  }

  // ---- normalize and scatter to ctx [N][E] ---------------------------------
#pragma unroll
  for (int r = 0; r < 8; ++r) {
    int m = m0 + r + 8 * hf;
    float inv = 1.0f / lrun[r];
    ctx16[(size_t)m * EE + h * DH + col]      = (_Float16)(acc0[r] * inv);
    ctx16[(size_t)m * EE + h * DH + 16 + col] = (_Float16)(acc1[r] * inv);
  }
}

// ---------------------------------------------------------------------------
// Kernel 3: out = ctx @ Wo^T + bo + x, then LayerNorm.
// grid.x = N/16 row tiles, 8 waves: wave w owns n-tiles 2w, 2w+1.
// A-fragments hoisted and reused across both tiles.
// ---------------------------------------------------------------------------
__global__ __launch_bounds__(256) void outln_kernel(
    const _Float16* __restrict__ ctx16, const _Float16* __restrict__ Wo16,
    const float* __restrict__ bo, const float* __restrict__ x,
    const float* __restrict__ gamma, const float* __restrict__ beta,
    float* __restrict__ out) {
  __shared__ float yb[16][EE];   // 16 KB

  const int m0 = blockIdx.x * 16;
  const int wv = threadIdx.x >> 5, lane = threadIdx.x & 31;
  const int col = lane & 15, hf = lane >> 4;

  v16h af[8];
  {
    const _Float16* arow = ctx16 + (size_t)(m0 + col) * EE;
#pragma unroll
    for (int k = 0; k < 8; ++k) af[k] = load_frag_rm(arow + 32 * k, hf);
  }

#pragma unroll
  for (int t = 0; t < 2; ++t) {
    int nt = wv * 2 + t;
    const _Float16* wrow = Wo16 + (size_t)(nt * 16 + col) * EE;
    v8f acc = {};
#pragma unroll
    for (int k = 0; k < 8; ++k)
      acc = wmma16(af[k], load_frag_rm(wrow + 32 * k, hf), acc);

    int cg = nt * 16 + col;
    float bb = bo[cg];
#pragma unroll
    for (int r = 0; r < 8; ++r) {
      int mloc = r + 8 * hf;
      yb[mloc][cg] = acc[r] + bb + x[(size_t)(m0 + mloc) * EE + cg];
    }
  }
  __syncthreads();

  // LayerNorm: 16 threads per row (a contiguous 16-lane group in each wave)
  const int row = threadIdx.x >> 4, c = threadIdx.x & 15;
  float s1 = 0.0f, s2 = 0.0f;
#pragma unroll
  for (int e = 0; e < 16; ++e) {
    float v = yb[row][c * 16 + e];
    s1 += v; s2 += v * v;
  }
  s1 = rsum16(s1);
  s2 = rsum16(s2);
  float mean = s1 * (1.0f / 256.0f);
  float var  = s2 * (1.0f / 256.0f) - mean * mean;
  float rstd = rsqrtf(var + 1e-5f);
  int m = m0 + row;
#pragma unroll
  for (int e = 0; e < 16; ++e) {
    int eg = c * 16 + e;
    out[(size_t)m * EE + eg] = (yb[row][eg] - mean) * rstd * gamma[eg] + beta[eg];
  }
}

// ---------------------------------------------------------------------------
// Host side
// ---------------------------------------------------------------------------
extern "C" void kernel_launch(void* const* d_in, const int* in_sizes, int n_in,
                              void* d_out, int out_size, void* d_ws, size_t ws_size,
                              hipStream_t stream) {
  (void)in_sizes; (void)n_in; (void)out_size; (void)ws_size;

  const float*     x     = (const float*)d_in[0];
  const float*     pe    = (const float*)d_in[1];
  const long long* xb    = (const long long*)d_in[2];
  const float*     Wq    = (const float*)d_in[3];
  const float*     Wk    = (const float*)d_in[4];
  const float*     Wv    = (const float*)d_in[5];
  const float*     bq    = (const float*)d_in[6];
  const float*     bk    = (const float*)d_in[7];
  const float*     bv    = (const float*)d_in[8];
  const float*     Wo    = (const float*)d_in[9];
  const float*     bo    = (const float*)d_in[10];
  const float*     gamma = (const float*)d_in[11];
  const float*     beta  = (const float*)d_in[12];
  float*           out   = (float*)d_out;

  char* ws = (char*)d_ws;
  const size_t NE2  = (size_t)NN * EE * sizeof(_Float16);   // 2 MB
  const size_t WE2  = (size_t)EE * EE * sizeof(_Float16);   // 128 KB
  _Float16* xp16  = (_Float16*)(ws);
  _Float16* x16   = (_Float16*)(ws + NE2);
  _Float16* Wq16  = (_Float16*)(ws + 2 * NE2);
  _Float16* Wk16  = (_Float16*)(ws + 2 * NE2 + WE2);
  _Float16* Wv16  = (_Float16*)(ws + 2 * NE2 + 2 * WE2);
  _Float16* Wo16  = (_Float16*)(ws + 2 * NE2 + 3 * WE2);
  _Float16* Q16   = (_Float16*)(ws + 2 * NE2 + 4 * WE2);
  _Float16* K16   = (_Float16*)(ws + 3 * NE2 + 4 * WE2);
  _Float16* V16   = (_Float16*)(ws + 4 * NE2 + 4 * WE2);
  _Float16* ctx16 = (_Float16*)(ws + 5 * NE2 + 4 * WE2);

  const int prep_blocks = (NN * EE + 4 * EE * EE + 255) / 256;
  prep_kernel<<<prep_blocks, 256, 0, stream>>>(x, pe, Wq, Wk, Wv, Wo,
                                               xp16, x16, Wq16, Wk16, Wv16, Wo16);

  proj_kernel<<<NN / 16, 256, 0, stream>>>(xp16, x16, Wq16, Wk16, Wv16,
                                           bq, bk, bv, Q16, K16, V16);

  attn_kernel<<<dim3(NN / 16, HH), 32, 0, stream>>>(Q16, K16, V16, xb, ctx16);

  outln_kernel<<<NN / 16, 256, 0, stream>>>(ctx16, Wo16, bo, x, gamma, beta, out);
}
